// KimiLinearAttention_79328045957802
// MI455X (gfx1250) — matrix operand order repaired and checked by reading, not compiled, not run on hardware
//
#include <hip/hip_runtime.h>

#define D_MODEL 1024
#define STATE_N 1024
#define RANK_N  64
#define CHUNK_N 64
#define HD      64

typedef __attribute__((ext_vector_type(16))) __bf16 v16bf;
typedef __attribute__((ext_vector_type(8)))  float  v8f;
typedef __bf16 bf16;

// ---------------------------------------------------------------------------
// Fragment loaders following CDNA5 ISA 7.12.2 (16-bit A 16x32, B 32x16).
// A: lane<16 -> K {k0..k0+7, k0+16..k0+23}; lane>=16 -> +8 on both halves.
// B: lane&15 = column N; same K order as A (lane<16 -> K {k0..k0+7, k0+16..k0+23}; lane>=16 -> +8).
// ---------------------------------------------------------------------------
__device__ __forceinline__ v16bf load_fragA(const bf16* __restrict__ A, int ld,
                                            int row0, int k0, int lane) {
  int r  = row0 + (lane & 15);
  int ks = k0 + ((lane >> 4) << 3);
  const bf16* p = A + (size_t)r * ld + ks;
  union { uint4 q[2]; v16bf v; } u;
  u.q[0] = *(const uint4*)(p);
  u.q[1] = *(const uint4*)(p + 16);
  return u.v;
}

__device__ __forceinline__ v16bf load_fragB(const bf16* __restrict__ W, int ld,
                                            int col0, int k0, int lane) {
  int n  = col0 + (lane & 15);
  int ks = k0 + ((lane >> 4) << 3);
  const bf16* p = W + (size_t)n * ld + ks;
  union { uint4 q[2]; v16bf v; } u;
  u.q[0] = *(const uint4*)(p);
  u.q[1] = *(const uint4*)(p + 16);
  return u.v;
}

// ---------------------------------------------------------------------------
// Epilogue modes
// ---------------------------------------------------------------------------
enum { EP_F32 = 0, EP_BF16 = 1, EP_ROPE = 2, EP_SIGMUL = 3, EP_ADD = 4 };

template <int EP>
__device__ __forceinline__ void epilogue_store(float val, int m, int n, int N, int T,
                                               const float* __restrict__ rope,
                                               const bf16* __restrict__ aux,
                                               bf16* __restrict__ outB,
                                               float* __restrict__ outF) {
  size_t idx = (size_t)m * N + n;
  if constexpr (EP == EP_F32) {
    outF[idx] = val;
  } else if constexpr (EP == EP_BF16) {
    outB[idx] = (bf16)val;
  } else if constexpr (EP == EP_ROPE) {
    int t = m % T;                      // row = b*T + t
    val *= rope[t * HD + (n & (HD - 1))];
    outB[idx] = (bf16)val;
  } else if constexpr (EP == EP_SIGMUL) {
    float g = 1.0f / (1.0f + __expf(-val));
    outB[idx] = (bf16)(g * (float)aux[idx]);   // aux may alias outB (same elem)
  } else {                                      // EP_ADD
    val += (float)aux[idx];
    outB[idx] = (bf16)val;
  }
}

// ---------------------------------------------------------------------------
// Generic C = A[M,K] @ W[N,K]^T, bf16 inputs, f32 accumulate via WMMA.
// Wave tile WM x WN, block = WAVES_M x WAVES_N waves (wave32).
// ---------------------------------------------------------------------------
template <int WM, int WN, int WAVES_M, int WAVES_N, int EP>
__global__ __launch_bounds__(WAVES_M* WAVES_N * 32) void gemm_bf16_wmma(
    const bf16* __restrict__ A, const bf16* __restrict__ W, int M, int N, int K,
    int T, const float* __restrict__ rope, const bf16* __restrict__ aux,
    bf16* __restrict__ outB, float* __restrict__ outF) {
  constexpr int MI = WM / 16, NI = WN / 16;
  const int lane  = threadIdx.x & 31;
  const int wave  = threadIdx.x >> 5;
  const int waveM = wave % WAVES_M;
  const int waveN = wave / WAVES_M;
  const int m0 = blockIdx.x * (WAVES_M * WM) + waveM * WM;
  const int n0 = blockIdx.y * (WAVES_N * WN) + waveN * WN;

  const v8f vzero = {0.f, 0.f, 0.f, 0.f, 0.f, 0.f, 0.f, 0.f};
  v8f acc[MI][NI];
#pragma unroll
  for (int i = 0; i < MI; ++i)
#pragma unroll
    for (int j = 0; j < NI; ++j) acc[i][j] = vzero;

  for (int k0 = 0; k0 < K; k0 += 32) {
    v16bf af[MI], bfr[NI];
#pragma unroll
    for (int i = 0; i < MI; ++i) af[i] = load_fragA(A, K, m0 + i * 16, k0, lane);
#pragma unroll
    for (int j = 0; j < NI; ++j) bfr[j] = load_fragB(W, K, n0 + j * 16, k0, lane);
    if (k0 + 32 < K) {  // prefetch next A slab (global_prefetch_b8)
      __builtin_prefetch(A + (size_t)(m0 + (lane & 15)) * K + k0 + 32, 0, 1);
    }
#pragma unroll
    for (int i = 0; i < MI; ++i)
#pragma unroll
      for (int j = 0; j < NI; ++j)
        acc[i][j] = __builtin_amdgcn_wmma_f32_16x16x32_bf16(
            false, af[i], false, bfr[j], (short)0, acc[i][j], false, false);
  }

  const int rsel = (lane >> 4) << 3;  // +8 rows for upper half-wave
  const int ncol = lane & 15;
#pragma unroll
  for (int i = 0; i < MI; ++i)
#pragma unroll
    for (int j = 0; j < NI; ++j)
#pragma unroll
      for (int r = 0; r < 8; ++r) {
        int m = m0 + i * 16 + r + rsel;
        int n = n0 + j * 16 + ncol;
        epilogue_store<EP>(acc[i][j][r], m, n, N, T, rope, aux, outB, outF);
      }
}

// ---------------------------------------------------------------------------
// Elementwise helpers
// ---------------------------------------------------------------------------
__global__ void k_cvt_bf16(const float* __restrict__ in, bf16* __restrict__ out,
                           size_t n) {
  size_t i = (size_t)blockIdx.x * blockDim.x + threadIdx.x;
  size_t st = (size_t)gridDim.x * blockDim.x;
  for (; i < n; i += st) out[i] = (bf16)in[i];
}

// in[rows][cols] -> out[cols][rows] with f32->bf16
__global__ void k_transpose_cvt(const float* __restrict__ in, bf16* __restrict__ out,
                                int rows, int cols) {
  size_t n = (size_t)rows * cols;
  size_t i = (size_t)blockIdx.x * blockDim.x + threadIdx.x;
  size_t st = (size_t)gridDim.x * blockDim.x;
  for (; i < n; i += st) {
    int r = (int)(i / cols), c = (int)(i % cols);
    out[(size_t)c * rows + r] = (bf16)in[i];
  }
}

__global__ void k_rope(const float* __restrict__ cs, float* __restrict__ rope,
                       size_t n) {  // n = T*HD ; rope = cos + sin
  size_t i = (size_t)blockIdx.x * blockDim.x + threadIdx.x;
  size_t st = (size_t)gridDim.x * blockDim.x;
  for (; i < n; i += st) rope[i] = cs[i] + cs[n + i];
}

__device__ __forceinline__ float decay_from(float d) {
  float sp = (d > 20.f) ? d : log1pf(expf(d));  // softplus
  return fminf(fmaxf(sp, 0.f), 1.f);
}

// ---------------------------------------------------------------------------
// Scan: s[t] = decay*s[t-1] + h[t] (equivalent to the reference's chunked
// cumsum formulation, without the unstable division by decay^t).
// Pass1: per-(b,chunk,channel) local terminal value with zero incoming state.
// Pass2: cross-chunk recurrence (factor decay^CHUNK); emits per-chunk incoming
//        state and the final state (second output of the reference).
// Pass3: replay each chunk with known incoming state, emit s as bf16.
// ---------------------------------------------------------------------------
__global__ void scan_pass1(const bf16* __restrict__ h, const float* __restrict__ diag,
                           float* __restrict__ localFinal, int T, int S, int nC,
                           int total) {
  int idx = blockIdx.x * blockDim.x + threadIdx.x;
  if (idx >= total) return;
  int s = idx % S;
  int c = (idx / S) % nC;
  int b = idx / (S * nC);
  float d = decay_from(diag[s]);
  const bf16* hp = h + ((size_t)(b * T + c * CHUNK_N)) * S + s;
  float st = 0.f;
#pragma unroll 4
  for (int t = 0; t < CHUNK_N; ++t) st = fmaf(d, st, (float)hp[(size_t)t * S]);
  localFinal[idx] = st;
}

__global__ void scan_pass2(const float* __restrict__ localFinal,
                           const float* __restrict__ diag,
                           float* __restrict__ stateIn, float* __restrict__ stateOut,
                           int S, int nC, int total) {
  int idx = blockIdx.x * blockDim.x + threadIdx.x;  // over B*S
  if (idx >= total) return;
  int s = idx % S;
  int b = idx / S;
  float d = decay_from(diag[s]);
  float d2 = d * d, d4 = d2 * d2, d8 = d4 * d4, d16 = d8 * d8, d32 = d16 * d16;
  float dC = d32 * d32;  // decay^64
  float st = 0.f;
  for (int c = 0; c < nC; ++c) {
    size_t q = ((size_t)b * nC + c) * S + s;
    stateIn[q] = st;
    st = fmaf(dC, st, localFinal[q]);
  }
  stateOut[(size_t)b * S + s] = st;  // state_f
}

__global__ void scan_pass3(const bf16* __restrict__ h, const float* __restrict__ diag,
                           const float* __restrict__ stateIn, bf16* __restrict__ sb,
                           int T, int S, int nC, int total) {
  int idx = blockIdx.x * blockDim.x + threadIdx.x;
  if (idx >= total) return;
  int s = idx % S;
  int c = (idx / S) % nC;
  int b = idx / (S * nC);
  float d = decay_from(diag[s]);
  size_t base = ((size_t)(b * T + c * CHUNK_N)) * S + s;
  float st = stateIn[idx];
#pragma unroll 4
  for (int t = 0; t < CHUNK_N; ++t) {
    st = fmaf(d, st, (float)h[base + (size_t)t * S]);
    sb[base + (size_t)t * S] = (bf16)st;
  }
}

// ---------------------------------------------------------------------------
// Host orchestration
// ---------------------------------------------------------------------------
extern "C" void kernel_launch(void* const* d_in, const int* in_sizes, int n_in,
                              void* d_out, int out_size, void* d_ws, size_t ws_size,
                              hipStream_t stream) {
  const float* x    = (const float*)d_in[0];
  const float* cs   = (const float*)d_in[1];
  const float* U    = (const float*)d_in[2];
  const float* V    = (const float*)d_in[3];
  const float* diag = (const float*)d_in[4];
  const float* Wg   = (const float*)d_in[5];
  const float* Wk   = (const float*)d_in[6];
  const float* Wv   = (const float*)d_in[7];
  const float* Wo   = (const float*)d_in[8];
  float* out = (float*)d_out;

  const int D = D_MODEL, S = STATE_N, R = RANK_N;
  const int T  = in_sizes[1] / (2 * HD);
  const int BT = in_sizes[0] / D;  // B*T rows
  const int B  = BT / T;
  const int nC = T / CHUNK_N;
  const size_t BTD = (size_t)BT * D;

  // Workspace layout (256B aligned, manual lifetime-based aliasing).
  char* ws = (char*)d_ws;
  size_t off = 0;
  auto alloc = [&](size_t bytes) -> char* {
    char* p = ws + off;
    off += (bytes + 255) & ~(size_t)255;
    return p;
  };
  bf16*  xb   = (bf16*)alloc(BTD * 2);             // x bf16; reused as outb later
  bf16*  Wkb  = (bf16*)alloc((size_t)D * D * 2);
  bf16*  Wvb  = (bf16*)alloc((size_t)D * D * 2);
  bf16*  Wgb  = (bf16*)alloc((size_t)S * D * 2);
  bf16*  Wob  = (bf16*)alloc((size_t)D * S * 2);
  bf16*  Ub   = (bf16*)alloc((size_t)S * R * 2);
  bf16*  Vtb  = (bf16*)alloc((size_t)R * D * 2);   // V^T
  float* rope = (float*)alloc((size_t)T * HD * 4);
  bf16*  kb   = (bf16*)alloc(BTD * 2);             // k_rope; reused as sb later
  bf16*  vb   = (bf16*)alloc(BTD * 2);             // v_rope
  bf16*  vtkb = (bf16*)alloc((size_t)BT * R * 2);
  bf16*  hb   = (bf16*)alloc(BTD * 2);             // delta then h
  float* lf   = (float*)alloc((size_t)B * nC * S * 4);
  float* sIn  = (float*)alloc((size_t)B * nC * S * 4);
  bf16*  sb   = kb;   // k dead after vtk GEMM
  bf16*  outb = xb;   // x dead after g GEMM

  const int TPB = 256;
  auto blocks = [](size_t n, int t) { return (unsigned)((n + t - 1) / t); };

  // Stage 0: precision conversion + rope table
  k_cvt_bf16<<<blocks(BTD, TPB), TPB, 0, stream>>>(x, xb, BTD);
  k_cvt_bf16<<<blocks((size_t)D * D, TPB), TPB, 0, stream>>>(Wk, Wkb, (size_t)D * D);
  k_cvt_bf16<<<blocks((size_t)D * D, TPB), TPB, 0, stream>>>(Wv, Wvb, (size_t)D * D);
  k_cvt_bf16<<<blocks((size_t)S * D, TPB), TPB, 0, stream>>>(Wg, Wgb, (size_t)S * D);
  k_cvt_bf16<<<blocks((size_t)D * S, TPB), TPB, 0, stream>>>(Wo, Wob, (size_t)D * S);
  k_cvt_bf16<<<blocks((size_t)S * R, TPB), TPB, 0, stream>>>(U, Ub, (size_t)S * R);
  k_transpose_cvt<<<blocks((size_t)D * R, TPB), TPB, 0, stream>>>(V, Vtb, D, R);
  k_rope<<<blocks((size_t)T * HD, TPB), TPB, 0, stream>>>(cs, rope, (size_t)T * HD);

  // GEMM configs: wave tile 32x64; blockA tile 64x256 (2x4 waves), blockB 256x64.
  dim3 blk(256);
  dim3 gridA(BT / 64, D / 256);      // N = 1024
  dim3 gridB(BT / 256, R / 64);      // N = 64

  // Stage 1/2: k = rope*(x@Wk^T), v = rope*(x@Wv^T)
  gemm_bf16_wmma<32, 64, 2, 4, EP_ROPE><<<gridA, blk, 0, stream>>>(
      xb, Wkb, BT, D, D, T, rope, nullptr, kb, nullptr);
  gemm_bf16_wmma<32, 64, 2, 4, EP_ROPE><<<gridA, blk, 0, stream>>>(
      xb, Wvb, BT, D, D, T, rope, nullptr, vb, nullptr);

  // Stage 3: vtk = k_rope @ V   (W = V^T, N=R)
  gemm_bf16_wmma<32, 64, 8, 1, EP_BF16><<<gridB, blk, 0, stream>>>(
      kb, Vtb, BT, R, D, T, nullptr, nullptr, vtkb, nullptr);

  // Stage 4: delta = vtk @ U^T  (K = R = 64)
  gemm_bf16_wmma<32, 64, 2, 4, EP_BF16><<<gridA, blk, 0, stream>>>(
      vtkb, Ub, BT, S, R, T, nullptr, nullptr, hb, nullptr);

  // Stage 5: h = sigmoid(x @ Wg^T) * delta   (in-place on hb)
  gemm_bf16_wmma<32, 64, 2, 4, EP_SIGMUL><<<gridA, blk, 0, stream>>>(
      xb, Wgb, BT, S, D, T, nullptr, hb, hb, nullptr);

  // Stage 6: chunked parallel scan -> s (bf16) and final state (f32, d_out tail)
  int totC = B * nC * S, totS = B * S;
  scan_pass1<<<blocks(totC, TPB), TPB, 0, stream>>>(hb, diag, lf, T, S, nC, totC);
  scan_pass2<<<blocks(totS, TPB), TPB, 0, stream>>>(lf, diag, sIn, out + BTD, S, nC, totS);
  scan_pass3<<<blocks(totC, TPB), TPB, 0, stream>>>(hb, diag, sIn, sb, T, S, nC, totC);

  // Stage 7: out_chunk = s @ Wo^T + v_rope
  gemm_bf16_wmma<32, 64, 2, 4, EP_ADD><<<gridA, blk, 0, stream>>>(
      sb, Wob, BT, D, S, T, nullptr, vb, outb, nullptr);

  // Stage 8: final = out_chunk @ Wo^T  -> f32 d_out
  gemm_bf16_wmma<32, 64, 2, 4, EP_F32><<<gridA, blk, 0, stream>>>(
      outb, Wob, BT, D, D, T, nullptr, nullptr, nullptr, out);
}
